// MC_E_GCL_33621003993975
// MI455X (gfx1250) — compile-verified, matrix operand order checked by reading, not run on hardware
//
#include <hip/hip_runtime.h>

#define HID 128
#define ETILE 64
#define NTILE 16

typedef __attribute__((ext_vector_type(16))) _Float16 v16h;
typedef __attribute__((ext_vector_type(8)))  _Float16 v8h;
typedef __attribute__((ext_vector_type(8)))  float    v8f;
typedef _Float16 half_t;

__device__ __forceinline__ float silu_f(float x) {
    return x / (1.0f + __expf(-x));
}

// Fragment-order index for a 16xK (A) or Kx16-per-column (B) f16 WMMA operand.
// Layout: [ktile][half(0/1)][m (0..M-1)][j (0..15)] halfs, 32B per (ktile,half,m).
// Lane half h covers K in {8h..8h+7} U {16+8h..16+8h+7} within a 32-wide ktile.
__device__ __forceinline__ int frag_idx(int k, int m, int M) {
    int kk  = k >> 5;
    int kr  = k & 31;
    int hh  = (kr >> 3) & 1;
    int kr2 = kr - 8 * hh;          // in {0..7} U {16..23}
    int j   = (kr2 < 8) ? kr2 : (kr2 - 8);
    return ((kk * 2 + hh) * M + m) * 16 + j;
}

__device__ __forceinline__ v8f wmma_f16(v16h a, v16h b, v8f c) {
    return __builtin_amdgcn_wmma_f32_16x16x32_f16(false, a, false, b,
                                                  (short)0, c, false, false);
}

__device__ __forceinline__ v8h pack8(float4 a, float4 b) {
    v8h r;
    r[0] = (_Float16)a.x; r[1] = (_Float16)a.y;
    r[2] = (_Float16)a.z; r[3] = (_Float16)a.w;
    r[4] = (_Float16)b.x; r[5] = (_Float16)b.y;
    r[6] = (_Float16)b.z; r[7] = (_Float16)b.w;
    return r;
}

// ---------------- small helper kernels ----------------

__global__ void k_zero(float* p, int n) {
    int i = blockIdx.x * blockDim.x + threadIdx.x;
    if (i < n) p[i] = 0.0f;
}

// f32 weight [K][nout] (row-major) -> f16 fragment-order [ktile][h][n][j],
// zero-padding columns nout..ncols-1.
__global__ void k_swz(const float* __restrict__ src, half_t* __restrict__ dst,
                      int K, int nout, int ncols) {
    int i = blockIdx.x * blockDim.x + threadIdx.x;
    if (i >= K * ncols) return;
    int k = i / ncols, n = i - k * ncols;
    float v = (n < nout) ? src[k * nout + n] : 0.0f;
    dst[frag_idx(k, n, ncols)] = (half_t)v;
}

// Per-edge geometry: sum-of-squares of radial (gram) and cdist features
// (32 cols, [c*8+f] radial / [c*8+4+f] dist, matching concat+reshape order),
// plus per-row-node edge counts.
__global__ void k_geom(const float* __restrict__ coord, const int* __restrict__ row,
                       const int* __restrict__ col, float* __restrict__ ss,
                       float* __restrict__ cnt, int E) {
    float acc[32];
#pragma unroll
    for (int i = 0; i < 32; i++) acc[i] = 0.0f;

    for (int e = blockIdx.x * blockDim.x + threadIdx.x; e < E;
         e += gridDim.x * blockDim.x) {
        int ri = row[e], cn = col[e];
        float ci[12], cj[12], d[12];
#pragma unroll
        for (int i = 0; i < 12; i++) {
            ci[i] = coord[ri * 12 + i];
            cj[i] = coord[cn * 12 + i];
            d[i]  = ci[i] - cj[i];
        }
        atomicAdd(&cnt[ri], 1.0f);
#pragma unroll
        for (int c = 0; c < 4; c++) {
#pragma unroll
            for (int f = 0; f < 4; f++) {
                float rv = d[c*3]*d[f*3] + d[c*3+1]*d[f*3+1] + d[c*3+2]*d[f*3+2];
                acc[c*8 + f] += rv * rv;
                float dx = ci[c*3]   - cj[f*3];
                float dy = ci[c*3+1] - cj[f*3+1];
                float dz = ci[c*3+2] - cj[f*3+2];
                acc[c*8 + 4 + f] += dx*dx + dy*dy + dz*dz;   // dist^2
            }
        }
    }
#pragma unroll
    for (int i = 0; i < 32; i++) {
#pragma unroll
        for (int off = 16; off > 0; off >>= 1)
            acc[i] += __shfl_down(acc[i], off, 32);
    }
    if ((threadIdx.x & 31) == 0) {
#pragma unroll
        for (int i = 0; i < 32; i++) atomicAdd(&ss[i], acc[i]);
    }
}

__global__ void k_norm(const float* __restrict__ ss, float* __restrict__ invn) {
    int i = threadIdx.x;
    if (i < 32) invn[i] = 1.0f / fmaxf(sqrtf(ss[i]), 1e-12f);
}

// ---------------- fused edge pipeline (WMMA) ----------------

__global__ void __launch_bounds__(256)
k_edge(const float* __restrict__ h, const float* __restrict__ coord,
       const int* __restrict__ row, const int* __restrict__ col,
       const half_t* __restrict__ w1s, const float* __restrict__ b1,
       const half_t* __restrict__ w2s, const float* __restrict__ b2,
       const half_t* __restrict__ cw1s, const float* __restrict__ cb1,
       const half_t* __restrict__ cwouts,
       const float* __restrict__ invn,
       float* __restrict__ nagg, float* __restrict__ agg, int E) {
    __shared__ __align__(32) half_t XA[9 * 2 * ETILE * 16];   // 64x288, frag order
    __shared__ __align__(32) half_t YA[4 * 2 * ETILE * 16];   // 64x128 ping
    __shared__ __align__(32) half_t YB[4 * 2 * ETILE * 16];   // 64x128 pong
    __shared__ int   s_row[ETILE];
    __shared__ int   s_col[ETILE];
    __shared__ float s_diff[ETILE][12];
    __shared__ int   s_nv;

    const int tid  = threadIdx.x;
    const int base = blockIdx.x * ETILE;
    if (tid == 0) s_nv = min(ETILE, E - base);
    if (tid < ETILE) {
        int e = base + tid; if (e >= E) e = E - 1;
        s_row[tid] = row[e];
    } else if (tid < 2 * ETILE) {
        int t2 = tid - ETILE;
        int e = base + t2; if (e >= E) e = E - 1;
        s_col[t2] = col[e];
    }
    __syncthreads();
    const int nv = s_nv;

    // ---- stage h[row] | h[col] into columns 0..255 of XA ----
    // 64 edges x 16 chunks of 16 floats; each 16-chunk packs to two b128 stores.
#pragma unroll
    for (int q = 0; q < 4; q++) {
        int chunk = q * 256 + tid;               // 0..1023
        int e   = chunk >> 4;
        int seg = chunk & 15;
        int node = (seg < 8) ? s_row[e] : s_col[e];
        int cb   = (seg & 7) * 16;
        const float4* hp = (const float4*)(h + (size_t)node * HID + cb);
        float4 v0 = hp[0], v1 = hp[1], v2 = hp[2], v3 = hp[3];
        int kbase = ((seg < 8) ? 0 : 128) + cb;  // multiple of 16
        int kk    = kbase >> 5;
        int joff  = (kbase & 16) ? 8 : 0;
        // k = kbase..kbase+7 -> half 0 ; kbase+8..kbase+15 -> half 1
        *(v8h*)(XA + ((kk * 2 + 0) * ETILE + e) * 16 + joff) = pack8(v0, v1);
        *(v8h*)(XA + ((kk * 2 + 1) * ETILE + e) * 16 + joff) = pack8(v2, v3);
    }
    // ---- normalized radial/dist features into columns 256..287 (ktile 8) ----
    if (tid < ETILE) {
        int ri = s_row[tid], cn = s_col[tid];
        float ci[12], cj[12], d[12], rr[32];
#pragma unroll
        for (int i = 0; i < 12; i++) {
            ci[i] = coord[ri * 12 + i];
            cj[i] = coord[cn * 12 + i];
            d[i]  = ci[i] - cj[i];
            s_diff[tid][i] = d[i];
        }
#pragma unroll
        for (int c = 0; c < 4; c++) {
#pragma unroll
            for (int f = 0; f < 4; f++) {
                float rv = d[c*3]*d[f*3] + d[c*3+1]*d[f*3+1] + d[c*3+2]*d[f*3+2];
                float dx = ci[c*3]   - cj[f*3];
                float dy = ci[c*3+1] - cj[f*3+1];
                float dz = ci[c*3+2] - cj[f*3+2];
                float dv = sqrtf(dx*dx + dy*dy + dz*dz);
                int i0 = c*8 + f, i1 = c*8 + 4 + f;
                rr[i0] = rv * invn[i0];
                rr[i1] = dv * invn[i1];
            }
        }
        // kr 0..7 -> (h0,j0..7); 8..15 -> (h1,j0..7); 16..23 -> (h0,j8..15); 24..31 -> (h1,j8..15)
#pragma unroll
        for (int g = 0; g < 4; g++) {
            v8h p;
#pragma unroll
            for (int j = 0; j < 8; j++) p[j] = (_Float16)rr[g * 8 + j];
            *(v8h*)(XA + ((8 * 2 + (g & 1)) * ETILE + tid) * 16 + (g >> 1) * 8) = p;
        }
    }
    __syncthreads();

    const int wv   = tid >> 5;
    const int lane = tid & 31;
    const int hh   = lane >> 4;
    const int lm   = lane & 15;
    const int nl   = wv * 16 + lm;      // output column 0..127

    // ---- GEMM1: X(64x288) @ e_w1 + b1, SiLU -> YA ----
    {
        v8f acc[4];
        float bv = b1[nl];
#pragma unroll
        for (int mt = 0; mt < 4; mt++)
#pragma unroll
            for (int r = 0; r < 8; r++) acc[mt][r] = bv;
        for (int kk = 0; kk < 9; kk++) {
            v16h bf = *(const v16h*)(w1s + ((kk*2 + hh) * HID + nl) * 16);
#pragma unroll
            for (int mt = 0; mt < 4; mt++) {
                v16h a = *(const v16h*)(XA + ((kk*2 + hh) * ETILE + mt*16 + lm) * 16);
                acc[mt] = wmma_f16(a, bf, acc[mt]);
            }
        }
#pragma unroll
        for (int mt = 0; mt < 4; mt++)
#pragma unroll
            for (int r = 0; r < 8; r++) {
                int m0 = mt * 16 + r + 8 * hh;
                YA[frag_idx(nl, m0, ETILE)] = (half_t)silu_f(acc[mt][r]);
            }
    }
    __syncthreads();

    // ---- GEMM2: Y1 @ e_w2 + b2, SiLU -> m; atomics to nagg; m -> YB ----
    {
        v8f acc[4];
        float bv = b2[nl];
#pragma unroll
        for (int mt = 0; mt < 4; mt++)
#pragma unroll
            for (int r = 0; r < 8; r++) acc[mt][r] = bv;
        for (int kk = 0; kk < 4; kk++) {
            v16h bf = *(const v16h*)(w2s + ((kk*2 + hh) * HID + nl) * 16);
#pragma unroll
            for (int mt = 0; mt < 4; mt++) {
                v16h a = *(const v16h*)(YA + ((kk*2 + hh) * ETILE + mt*16 + lm) * 16);
                acc[mt] = wmma_f16(a, bf, acc[mt]);
            }
        }
#pragma unroll
        for (int mt = 0; mt < 4; mt++)
#pragma unroll
            for (int r = 0; r < 8; r++) {
                int m0 = mt * 16 + r + 8 * hh;
                float mv = silu_f(acc[mt][r]);
                YB[frag_idx(nl, m0, ETILE)] = (half_t)mv;
                if (m0 < nv)
                    atomicAdd(&nagg[(size_t)s_row[m0] * HID + nl], mv);
            }
    }
    __syncthreads();

    // ---- GEMM3: m @ c_w1 + c_b1, SiLU -> YA ----
    {
        v8f acc[4];
        float bv = cb1[nl];
#pragma unroll
        for (int mt = 0; mt < 4; mt++)
#pragma unroll
            for (int r = 0; r < 8; r++) acc[mt][r] = bv;
        for (int kk = 0; kk < 4; kk++) {
            v16h bf = *(const v16h*)(cw1s + ((kk*2 + hh) * HID + nl) * 16);
#pragma unroll
            for (int mt = 0; mt < 4; mt++) {
                v16h a = *(const v16h*)(YB + ((kk*2 + hh) * ETILE + mt*16 + lm) * 16);
                acc[mt] = wmma_f16(a, bf, acc[mt]);
            }
        }
#pragma unroll
        for (int mt = 0; mt < 4; mt++)
#pragma unroll
            for (int r = 0; r < 8; r++) {
                int m0 = mt * 16 + r + 8 * hh;
                YA[frag_idx(nl, m0, ETILE)] = (half_t)silu_f(acc[mt][r]);
            }
    }
    __syncthreads();

    // ---- GEMM4: Y3 @ c_wout (128x4 zero-padded to 16 cols); coord atomics ----
    if (wv < 4) {
        int mt = wv;   // wave w -> edges w*16..w*16+15
        v8f acc;
#pragma unroll
        for (int r = 0; r < 8; r++) acc[r] = 0.0f;
        for (int kk = 0; kk < 4; kk++) {
            v16h bf = *(const v16h*)(cwouts + ((kk*2 + hh) * 16 + lm) * 16);
            v16h a  = *(const v16h*)(YA + ((kk*2 + hh) * ETILE + mt*16 + lm) * 16);
            acc = wmma_f16(a, bf, acc);
        }
        int c = lm;
        if (c < 4) {
#pragma unroll
            for (int r = 0; r < 8; r++) {
                int m0 = mt * 16 + r + 8 * hh;
                if (m0 < nv) {
                    int node = s_row[m0];
                    float wvv = acc[r];
#pragma unroll
                    for (int dd = 0; dd < 3; dd++)
                        atomicAdd(&agg[(size_t)node * 12 + c*3 + dd],
                                  s_diff[m0][c*3 + dd] * wvv);
                }
            }
        }
    }
}

// ---------------- node MLP (WMMA) ----------------

__global__ void __launch_bounds__(256)
k_node(const float* __restrict__ h, const float* __restrict__ nagg,
       const half_t* __restrict__ w1s, const float* __restrict__ b1,
       const half_t* __restrict__ w2s, const float* __restrict__ b2,
       float* __restrict__ hout, int N) {
    __shared__ __align__(32) half_t XN[8 * 2 * NTILE * 16];  // 16x256
    __shared__ __align__(32) half_t YN[4 * 2 * NTILE * 16];  // 16x128

    const int tid  = threadIdx.x;
    const int base = blockIdx.x * NTILE;
    const int nv   = min(NTILE, N - base);

    {   // stage [h | nagg] -> XN ; 16 floats/thread -> two packed b128 stores
        int m   = tid >> 4;
        int seg = tid & 15;
        int node = base + ((m < nv) ? m : (nv - 1));
        const float* src = (seg < 8) ? (h    + (size_t)node * HID + (seg & 7) * 16)
                                     : (nagg + (size_t)node * HID + (seg & 7) * 16);
        const float4* sp = (const float4*)src;
        float4 v0 = sp[0], v1 = sp[1], v2 = sp[2], v3 = sp[3];
        int kbase = ((seg < 8) ? 0 : 128) + (seg & 7) * 16;
        int kk    = kbase >> 5;
        int joff  = (kbase & 16) ? 8 : 0;
        *(v8h*)(XN + ((kk * 2 + 0) * NTILE + m) * 16 + joff) = pack8(v0, v1);
        *(v8h*)(XN + ((kk * 2 + 1) * NTILE + m) * 16 + joff) = pack8(v2, v3);
    }
    __syncthreads();

    const int wv   = tid >> 5;
    const int lane = tid & 31;
    const int hh   = lane >> 4;
    const int lm   = lane & 15;
    const int nl   = wv * 16 + lm;

    {   // GEMM A: nin(16x256) @ n_w1 + b1, SiLU -> YN
        v8f acc;
        float bv = b1[nl];
#pragma unroll
        for (int r = 0; r < 8; r++) acc[r] = bv;
        for (int kk = 0; kk < 8; kk++) {
            v16h bf = *(const v16h*)(w1s + ((kk*2 + hh) * HID + nl) * 16);
            v16h a  = *(const v16h*)(XN + ((kk*2 + hh) * NTILE + lm) * 16);
            acc = wmma_f16(a, bf, acc);
        }
#pragma unroll
        for (int r = 0; r < 8; r++)
            YN[frag_idx(nl, r + 8 * hh, NTILE)] = (half_t)silu_f(acc[r]);
    }
    __syncthreads();

    {   // GEMM B: @ n_w2 + b2, + residual h -> hout
        v8f acc;
        float bv = b2[nl];
#pragma unroll
        for (int r = 0; r < 8; r++) acc[r] = bv;
        for (int kk = 0; kk < 4; kk++) {
            v16h bf = *(const v16h*)(w2s + ((kk*2 + hh) * HID + nl) * 16);
            v16h a  = *(const v16h*)(YN + ((kk*2 + hh) * NTILE + lm) * 16);
            acc = wmma_f16(a, bf, acc);
        }
#pragma unroll
        for (int r = 0; r < 8; r++) {
            int m0 = r + 8 * hh;
            if (m0 < nv) {
                size_t idx = (size_t)(base + m0) * HID + nl;
                hout[idx] = h[idx] + acc[r];
            }
        }
    }
}

__global__ void k_coord(const float* __restrict__ coord, const float* __restrict__ agg,
                        const float* __restrict__ cnt, float* __restrict__ out, int N) {
    int i = blockIdx.x * blockDim.x + threadIdx.x;
    if (i >= N * 12) return;
    int node = i / 12;
    out[i] = coord[i] + agg[i] / fmaxf(cnt[node], 1.0f);
}

// ---------------- launch ----------------

extern "C" void kernel_launch(void* const* d_in, const int* in_sizes, int n_in,
                              void* d_out, int out_size, void* d_ws, size_t ws_size,
                              hipStream_t stream) {
    const float* h      = (const float*)d_in[0];
    const float* coord  = (const float*)d_in[1];
    const int*   row    = (const int*)d_in[2];
    const int*   col    = (const int*)d_in[3];
    const float* e_w1   = (const float*)d_in[4];
    const float* e_b1   = (const float*)d_in[5];
    const float* e_w2   = (const float*)d_in[6];
    const float* e_b2   = (const float*)d_in[7];
    const float* c_w1   = (const float*)d_in[8];
    const float* c_b1   = (const float*)d_in[9];
    const float* c_wout = (const float*)d_in[10];
    const float* n_w1   = (const float*)d_in[11];
    const float* n_b1   = (const float*)d_in[12];
    const float* n_w2   = (const float*)d_in[13];
    const float* n_b2   = (const float*)d_in[14];

    const int E = in_sizes[2];
    const int N = in_sizes[0] / HID;

    // workspace layout (floats then f16 weights)
    float* nagg = (float*)d_ws;                 // N*128
    float* agg  = nagg + (size_t)N * HID;       // N*12
    float* cnt  = agg  + (size_t)N * 12;        // N
    float* ss   = cnt  + N;                     // 32
    float* invn = ss   + 32;                    // 32
    half_t* w1s    = (half_t*)(invn + 32);      // 9*2*128*16 = 36864
    half_t* w2s    = w1s    + 36864;            // 16384
    half_t* cw1s   = w2s    + 16384;            // 16384
    half_t* cwouts = cw1s   + 16384;            // 2048 (padded to 16 cols)
    half_t* nw1s   = cwouts + 2048;             // 32768
    half_t* nw2s   = nw1s   + 32768;            // 16384

    int zn = N * HID + N * 12 + N + 32;         // nagg|agg|cnt|ss contiguous
    k_zero<<<(zn + 255) / 256, 256, 0, stream>>>(nagg, zn);

    k_swz<<<(288 * 128 + 255) / 256, 256, 0, stream>>>(e_w1, w1s, 288, 128, 128);
    k_swz<<<(128 * 128 + 255) / 256, 256, 0, stream>>>(e_w2, w2s, 128, 128, 128);
    k_swz<<<(128 * 128 + 255) / 256, 256, 0, stream>>>(c_w1, cw1s, 128, 128, 128);
    k_swz<<<(128 * 16  + 255) / 256, 256, 0, stream>>>(c_wout, cwouts, 128, 4, 16);
    k_swz<<<(256 * 128 + 255) / 256, 256, 0, stream>>>(n_w1, nw1s, 256, 128, 128);
    k_swz<<<(128 * 128 + 255) / 256, 256, 0, stream>>>(n_w2, nw2s, 128, 128, 128);

    k_geom<<<1250, 256, 0, stream>>>(coord, row, col, ss, cnt, E);
    k_norm<<<1, 32, 0, stream>>>(ss, invn);

    int etiles = (E + ETILE - 1) / ETILE;
    k_edge<<<etiles, 256, 0, stream>>>(h, coord, row, col, w1s, e_b1, w2s, e_b2,
                                       cw1s, c_b1, cwouts, invn, nagg, agg, E);

    float* hout = (float*)d_out;
    float* cout = hout + (size_t)N * HID;
    int ntiles = (N + NTILE - 1) / NTILE;
    k_node<<<ntiles, 256, 0, stream>>>(h, nagg, nw1s, n_b1, nw2s, n_b2, hout, N);
    k_coord<<<(N * 12 + 255) / 256, 256, 0, stream>>>(coord, agg, cnt, cout, N);
}
